// SparseNet_4750233829844
// MI455X (gfx1250) — matrix operand order repaired and checked by reading, not compiled, not run on hardware
//
#include <hip/hip_runtime.h>
#include <stdint.h>

typedef __attribute__((ext_vector_type(16))) _Float16 v16h;
typedef __attribute__((ext_vector_type(8)))  _Float16 v8h;
typedef __attribute__((ext_vector_type(8)))  float    v8f;

#define GS   48
#define GV   (GS*GS*GS)        // 110592 voxels per batch (unpadded)
#define PD   50                // padded dim (1-voxel zero halo)
#define PV   (PD*PD*PD)        // 125000 padded voxels per batch
#define NB   2                 // batch
#define CH   32                // padded channel count (Cin padded 16->32 for layer 1)
#define CIN1 16
#define MAXN 5000

__device__ __forceinline__ long pidx(int b, int x, int y, int z) {
    // padded voxel index; (x,y,z) are unpadded coords in [-1, GS]
    return (long)b * PV + (((long)(x + 1) * PD) + (y + 1)) * PD + (z + 1);
}

// ---------------- utility fills ----------------
__global__ void fill_u32(uint32_t* __restrict__ p, uint32_t val, long n) {
    long i = (long)blockIdx.x * blockDim.x + threadIdx.x;
    if (i < n) p[i] = val;
}

// ---------------- weight pack: [3,3,3,Cin,32] f32 -> [27][32][32] f16 (cin padded) ----
__global__ void pack_weights(const float* __restrict__ w, _Float16* __restrict__ wp, int cin) {
    int i = blockIdx.x * blockDim.x + threadIdx.x;     // over 27*32*32
    if (i >= 27 * 32 * 32) return;
    int ci = i & 31;
    int co = (i >> 5) & 31;
    int t  = i >> 10;
    float v = (ci < cin) ? w[((long)t * cin + ci) * 32 + co] : 0.0f;
    wp[((long)t * 32 + co) * 32 + ci] = (_Float16)v;
}

// ---------------- scatter: deterministic last-point-wins via atomicMax winner ----------
__global__ void scatter_winner(const int* __restrict__ coords, int* __restrict__ winner,
                               int npts_total, int nper) {
    int gid = blockIdx.x * blockDim.x + threadIdx.x;
    if (gid >= npts_total) return;
    int b = gid / nper;
    const int* c = coords + (long)gid * 3;
    long vox = (long)b * GV + (((long)c[0] * GS + c[1]) * GS + c[2]);
    atomicMax(&winner[vox], gid);
}

__global__ void scatter_write(const float* __restrict__ feats, const int* __restrict__ coords,
                              const int* __restrict__ winner, _Float16* __restrict__ grid,
                              float* __restrict__ occ, int npts_total, int nper) {
    int gid = blockIdx.x * blockDim.x + threadIdx.x;
    if (gid >= npts_total) return;
    int b = gid / nper;
    const int* c = coords + (long)gid * 3;
    long vox = (long)b * GV + (((long)c[0] * GS + c[1]) * GS + c[2]);
    if (winner[vox] != gid) return;
    occ[vox] = 1.0f;
    const float* f = feats + (long)gid * CIN1;
    _Float16* g = grid + pidx(b, c[0], c[1], c[2]) * CH;
#pragma unroll
    for (int ci = 0; ci < CIN1; ++ci) g[ci] = (_Float16)f[ci];
}

// ---------------- 3x3x3 binary dilation of occupancy (unpadded masks) ---------------
__global__ void dilate_occ(const float* __restrict__ in, float* __restrict__ out) {
    long i = (long)blockIdx.x * blockDim.x + threadIdx.x;
    if (i >= (long)NB * GV) return;
    int v = (int)(i % GV);
    int b = (int)(i / GV);
    int z = v % GS, y = (v / GS) % GS, x = v / (GS * GS);
    float m = 0.0f;
    for (int dx = -1; dx <= 1; ++dx)
        for (int dy = -1; dy <= 1; ++dy)
            for (int dz = -1; dz <= 1; ++dz) {
                int xx = x + dx, yy = y + dy, zz = z + dz;
                if ((unsigned)xx < (unsigned)GS && (unsigned)yy < (unsigned)GS &&
                    (unsigned)zz < (unsigned)GS) {
                    float t = in[(long)b * GV + ((xx * GS + yy) * GS + zz)];
                    m = t > m ? t : m;
                }
            }
    out[i] = (m > 0.0f) ? 1.0f : 0.0f;
}

// ---------------- implicit-GEMM 3x3x3 conv via v_wmma_f32_16x16x32_f16 --------------
// Halo-padded input: no bounds checks, tap offsets are immediate offsets.
// One wave: 16 consecutive-z voxels x 32 out channels = 2 WMMAs/tap sharing one B load.
__global__ __launch_bounds__(256)
void conv3d_wmma(const _Float16* __restrict__ in, const _Float16* __restrict__ wp,
                 const float* __restrict__ mask, _Float16* __restrict__ out, int relu) {
    int wave = (blockIdx.x * blockDim.x + threadIdx.x) >> 5;
    int lane = threadIdx.x & 31;

    int tz = wave % 3;             int t1 = wave / 3;
    int y  = t1 % GS;              int t2 = t1 / GS;
    int x  = t2 % GS;              int b  = t2 / GS;

    int ln   = lane & 15;          // A: row M (cout); B/D: column N (voxel)
    int hi16 = lane >> 4;
    int z    = tz * 16 + ln;       // this lane's B/D voxel z

    // per-lane base pointers; all tap offsets below are compile-time immediates
    const _Float16* bptr  = in + pidx(b, x, y, z) * CH + hi16 * 16;
    const _Float16* aptr0 = wp + (long)ln * 32 + hi16 * 16;          // cout = ln
    const _Float16* aptr1 = wp + (long)(16 + ln) * 32 + hi16 * 16;   // cout = 16+ln

    v8f acc0 = {}, acc1 = {};
#pragma unroll
    for (int t = 0; t < 27; ++t) {
        const int dz = t % 3 - 1, dy = (t / 3) % 3 - 1, dx = t / 9 - 1;
        const int boff = ((dx * PD + dy) * PD + dz) * CH;   // constant per tap
        const int aoff = t * 32 * 32;                       // constant per tap

        v16h bmat = *(const v16h*)(bptr + boff);            // 32B contiguous, always valid

        v8h a0lo = *(const v8h*)(aptr0 + aoff);
        v8h a0hi = *(const v8h*)(aptr0 + aoff + 8);
        v8h a1lo = *(const v8h*)(aptr1 + aoff);
        v8h a1hi = *(const v8h*)(aptr1 + aoff + 8);
        v16h a0, a1;
#pragma unroll
        for (int i = 0; i < 8; ++i) {
            a0[i] = a0lo[i]; a0[i + 8] = a0hi[i];
            a1[i] = a1lo[i]; a1[i + 8] = a1hi[i];
        }

        acc0 = __builtin_amdgcn_wmma_f32_16x16x32_f16(false, a0, false, bmat,
                                                      (short)0, acc0, false, false);
        acc1 = __builtin_amdgcn_wmma_f32_16x16x32_f16(false, a1, false, bmat,
                                                      (short)0, acc1, false, false);
    }

    // D: acc[r] -> (M = 8*hi16 + r, N = ln); 8 consecutive cout -> one 16B store each
    float m = mask[(long)b * GV + ((x * GS + y) * GS + z)];
    _Float16* op = out + pidx(b, x, y, z) * CH;
    v8h o0, o1;
#pragma unroll
    for (int r = 0; r < 8; ++r) {
        float v0 = acc0[r], v1 = acc1[r];
        if (relu) { v0 = v0 < 0.0f ? 0.0f : v0; v1 = v1 < 0.0f ? 0.0f : v1; }
        o0[r] = (_Float16)(v0 * m);
        o1[r] = (_Float16)(v1 * m);
    }
    *(v8h*)(op + hi16 * 8)      = o0;   // cout 0..15 half
    *(v8h*)(op + 16 + hi16 * 8) = o1;   // cout 16..31 half
}

// ---------------- ordered compaction (lexicographic) of active voxels --------------
__global__ __launch_bounds__(1024)
void gather_active(const _Float16* __restrict__ grid, const float* __restrict__ occ,
                   float* __restrict__ out) {
    __shared__ int sh[1024];
    __shared__ int sbase;
    int b = blockIdx.x;
    int tid = threadIdx.x;
    if (tid == 0) sbase = 0;
    __syncthreads();
    for (int chunk = 0; chunk < GV; chunk += 1024) {
        int v = chunk + tid;
        int act = (v < GV) && (occ[(long)b * GV + v] > 0.0f);
        sh[tid] = act;
        __syncthreads();
        for (int off = 1; off < 1024; off <<= 1) {
            int t = (tid >= off) ? sh[tid - off] : 0;
            __syncthreads();
            sh[tid] += t;
            __syncthreads();
        }
        int pos = sbase + sh[tid] - act;   // exclusive prefix + running base
        int total = sh[1023];
        if (act && pos < MAXN) {
            int z = v % GS, y = (v / GS) % GS, x = v / (GS * GS);
            const _Float16* src = grid + pidx(b, x, y, z) * CH;
            float* dst = out + ((long)b * MAXN + pos) * CH;
#pragma unroll
            for (int c = 0; c < CH; ++c) dst[c] = (float)src[c];
        }
        __syncthreads();
        if (tid == 0) sbase += total;
        __syncthreads();
    }
}

// ------------------------------- launch ---------------------------------------------
extern "C" void kernel_launch(void* const* d_in, const int* in_sizes, int n_in,
                              void* d_out, int out_size, void* d_ws, size_t ws_size,
                              hipStream_t stream) {
    const float* features = (const float*)d_in[0];
    const int*   coords   = (const int*)d_in[1];
    const float* w[5] = { (const float*)d_in[2], (const float*)d_in[3],
                          (const float*)d_in[4], (const float*)d_in[5],
                          (const float*)d_in[6] };
    float* out = (float*)d_out;

    const int nper = in_sizes[0] / (NB * CIN1);   // points per batch (20000)
    const int npts = NB * nper;
    const long BV  = (long)NB * GV;               // unpadded voxels
    const long BPV = (long)NB * PV;               // padded voxels

    // workspace carve-up (256B aligned)
    size_t off = 0;
    auto carve = [&](size_t bytes) { size_t o = off; off = (off + bytes + 255) & ~(size_t)255; return o; };
    char* ws = (char*)d_ws;
    _Float16* grid0 = (_Float16*)(ws + carve(BPV * CH * sizeof(_Float16)));
    _Float16* grid1 = (_Float16*)(ws + carve(BPV * CH * sizeof(_Float16)));
    float* occ0 = (float*)(ws + carve(BV * sizeof(float)));
    float* occ1 = (float*)(ws + carve(BV * sizeof(float)));
    float* occ2 = (float*)(ws + carve(BV * sizeof(float)));
    int*   winner = (int*)(ws + carve(BV * sizeof(int)));
    _Float16* wp[5];
    for (int l = 0; l < 5; ++l)
        wp[l] = (_Float16*)(ws + carve(27 * 32 * 32 * sizeof(_Float16)));

    const int T = 256;
    auto blocks = [](long n, int t) { return (int)((n + t - 1) / t); };

    // init: zero both padded grids (halo determinism), occ0, d_out; winner = -1
    fill_u32<<<blocks(BPV * CH / 2, T), T, 0, stream>>>((uint32_t*)grid0, 0u, BPV * CH / 2);
    fill_u32<<<blocks(BPV * CH / 2, T), T, 0, stream>>>((uint32_t*)grid1, 0u, BPV * CH / 2);
    fill_u32<<<blocks(BV, T), T, 0, stream>>>((uint32_t*)occ0, 0u, BV);
    fill_u32<<<blocks(BV, T), T, 0, stream>>>((uint32_t*)winner, 0xFFFFFFFFu, BV);
    fill_u32<<<blocks(out_size, T), T, 0, stream>>>((uint32_t*)out, 0u, out_size);

    // pack weights to f16 [27][cout][cin32]
    pack_weights<<<blocks(27 * 1024, T), T, 0, stream>>>(w[0], wp[0], CIN1);
    for (int l = 1; l < 5; ++l)
        pack_weights<<<blocks(27 * 1024, T), T, 0, stream>>>(w[l], wp[l], 32);

    // scatter points (deterministic winner resolution for duplicate coords)
    scatter_winner<<<blocks(npts, T), T, 0, stream>>>(coords, winner, npts, nper);
    scatter_write<<<blocks(npts, T), T, 0, stream>>>(features, coords, winner, grid0, occ0,
                                                     npts, nper);

    // dilated active sets for the two SparseConv3d layers
    dilate_occ<<<blocks(BV, T), T, 0, stream>>>(occ0, occ1);
    dilate_occ<<<blocks(BV, T), T, 0, stream>>>(occ1, occ2);

    // conv stack: one wave per 16-voxel x 32-cout tile
    const long conv_threads = (long)NB * GS * GS * 3 * 32;
    const int  conv_blocks  = (int)(conv_threads / T);
    conv3d_wmma<<<conv_blocks, T, 0, stream>>>(grid0, wp[0], occ0, grid1, 1);  // SubM1+ReLU
    conv3d_wmma<<<conv_blocks, T, 0, stream>>>(grid1, wp[1], occ0, grid0, 1);  // SubM2+ReLU
    conv3d_wmma<<<conv_blocks, T, 0, stream>>>(grid0, wp[2], occ0, grid1, 1);  // SubM3+ReLU
    conv3d_wmma<<<conv_blocks, T, 0, stream>>>(grid1, wp[3], occ1, grid0, 1);  // Sparse4+ReLU
    conv3d_wmma<<<conv_blocks, T, 0, stream>>>(grid0, wp[4], occ2, grid1, 0);  // Sparse5

    // ordered per-batch compaction of active voxels -> [B, 5000, 32] f32
    gather_active<<<NB, 1024, 0, stream>>>(grid1, occ2, out);
}
